// RVQVAEBottleneck_23957327577860
// MI455X (gfx1250) — matrix operand order repaired and checked by reading, not compiled, not run on hardware
//
#include <hip/hip_runtime.h>
#include <hip/hip_bf16.h>

// ---------- problem constants ----------
#define BB   8
#define NN   4096
#define DD   64
#define KK   1024
#define QQ   8
#define ROWS (BB * NN)            // 32768
#define OUT_N   (BB * DD * NN)    // 2097152
#define KL_AT   OUT_N             // 1 float
#define IDX_AT  (OUT_N + 1)       // 262144 floats
#define LOSS_AT (OUT_N + 1 + BB * NN * QQ)

#define CHUNK_CODES 128           // codes staged in LDS per step (16 KB bf16)

// ---------- workspace byte offsets ----------
#define WS_RESID   ((size_t)0)                    // ROWS*DD f32   = 8 MB
#define WS_RESIDBF ((size_t)8388608)              // ROWS*DD bf16  = 4 MB
#define WS_OUTACC  ((size_t)12582912)             // ROWS*DD f32   = 8 MB
#define WS_CBBF    ((size_t)20971520)             // QQ*KK*DD bf16 = 1 MB
#define WS_C2      ((size_t)22020096)             // QQ*KK f32     = 32 KB
#define WS_SCAL    ((size_t)22052864)             // [0]=kl, [1..8]=loss[q]

typedef __attribute__((ext_vector_type(16))) __bf16 v16bf;
typedef __attribute__((ext_vector_type(8)))  float  v8f;
typedef __attribute__((ext_vector_type(4)))  unsigned int u32x4;

union Frag {
    v16bf bf;
    u32x4 u[2];
};

// ---------- small helpers ----------
__device__ __forceinline__ unsigned pcg_hash(unsigned v) {
    v = v * 747796405u + 2891336453u;
    unsigned w = ((v >> ((v >> 28u) + 4u)) ^ v) * 277803737u;
    return (w >> 22u) ^ w;
}

// ---------- prep: zero accumulators ----------
__global__ void k_zero(float* outacc, float* scal) {
    int gid = blockIdx.x * blockDim.x + threadIdx.x;
    if (gid < ROWS * DD) outacc[gid] = 0.0f;
    if (gid < 9) scal[gid] = 0.0f;
}

// ---------- prep: codebooks fp32 -> bf16 ----------
__global__ void k_cvt(const float* __restrict__ cb, __hip_bfloat16* __restrict__ cbbf) {
    int gid = blockIdx.x * blockDim.x + threadIdx.x;
    if (gid < QQ * KK * DD) cbbf[gid] = __float2bfloat16(cb[gid]);
}

// ---------- prep: c2[q][code] = sum_d cb^2 ----------
__global__ void k_c2(const float* __restrict__ cb, float* __restrict__ c2) {
    int gid = blockIdx.x * blockDim.x + threadIdx.x;   // 0..8191
    if (gid >= QQ * KK) return;
    const float* row = cb + (size_t)gid * DD;
    float s = 0.0f;
    #pragma unroll
    for (int d = 0; d < DD; ++d) { float v = row[d]; s += v * v; }
    c2[gid] = s;
}

// ---------- VAE sample + KL, write residual (f32 + bf16) in [row][d] ----------
__global__ void k_vae(const float* __restrict__ x, const int* __restrict__ seedp,
                      float* __restrict__ resid, __hip_bfloat16* __restrict__ residbf,
                      float* __restrict__ scal) {
    __shared__ float sm[256];
    int gid = blockIdx.x * blockDim.x + threadIdx.x;   // over BB*DD*NN
    float klv = 0.0f;
    if (gid < BB * DD * NN) {
        int n  = gid & (NN - 1);
        int ch = (gid >> 12) & (DD - 1);
        int b  = gid >> 18;
        float mean  = x[((size_t)(b * 128 + ch)      * NN) + n];
        float scale = x[((size_t)(b * 128 + DD + ch) * NN) + n];
        // softplus + eps
        float sp = (scale > 20.0f) ? scale : log1pf(expf(scale));
        float stdev = sp + 1e-4f;
        float var = stdev * stdev;
        float logvar = logf(var);
        // deterministic pseudo-gaussian noise (Box-Muller on hashed index)
        unsigned seed = (unsigned)seedp[0];
        unsigned h1 = pcg_hash((unsigned)gid ^ (seed * 0x9E3779B9u));
        unsigned h2 = pcg_hash((unsigned)gid * 0x85EBCA6Bu + seed + 0x165667B1u);
        float u1 = ((float)h1 + 0.5f) * (1.0f / 4294967296.0f);
        float u2 = ((float)h2 + 0.5f) * (1.0f / 4294967296.0f);
        float z = sqrtf(-2.0f * logf(u1)) * __cosf(6.283185307f * u2);
        float lat = z * stdev + mean;
        size_t row = (size_t)b * NN + n;               // [b, n, d] layout
        size_t off = row * DD + ch;
        resid[off]   = lat;
        residbf[off] = __float2bfloat16(lat);
        klv = mean * mean + var - logvar - 1.0f;
    }
    // block reduce KL
    sm[threadIdx.x] = klv;
    __syncthreads();
    for (int s = 128; s > 0; s >>= 1) {
        if ((int)threadIdx.x < s) sm[threadIdx.x] += sm[threadIdx.x + s];
        __syncthreads();
    }
    if (threadIdx.x == 0) atomicAdd(&scal[0], sm[0]);
}

// ---------- one RVQ round: LDS-staged WMMA nearest-neighbor + exact update ----
// grid 512 x 128 threads (4 waves); one wave = one 16-row tile; block shares the
// codebook through LDS in 16 KB chunks.
__global__ __launch_bounds__(128) void k_rvq(
                      const __hip_bfloat16* __restrict__ residbf,
                      float* __restrict__ resid,
                      __hip_bfloat16* __restrict__ residbf_w,
                      const __hip_bfloat16* __restrict__ cbbf,
                      const float* __restrict__ cbf32,
                      const float* __restrict__ c2,
                      float* __restrict__ outacc,
                      float* __restrict__ scal,
                      float* __restrict__ dout,
                      int q) {
    __shared__ __hip_bfloat16 scb[CHUNK_CODES * DD];   // 16 KB codebook chunk

    const int lane = threadIdx.x & 31;
    const int wave = threadIdx.x >> 5;
    const int tile = blockIdx.x * 4 + wave;            // 0..2047
    const int rowbase = tile * 16;
    const int half = (lane >> 4) & 1;                  // 0: lanes 0-15, 1: lanes 16-31
    const int lrow = lane & 15;

    // ---- A fragments: residual bf16, 16 rows x 64 K, ISA 16-bit A layout ----
    const char* rp = (const char*)residbf + ((size_t)(rowbase + lrow) * DD) * 2;
    Frag a0, a1;
    a0.u[0] = *(const u32x4*)(rp +       half * 16);   // K 0-7   / 8-15
    a0.u[1] = *(const u32x4*)(rp + 32  + half * 16);   // K 16-23 / 24-31
    a1.u[0] = *(const u32x4*)(rp + 64  + half * 16);   // K 32-39 / 40-47
    a1.u[1] = *(const u32x4*)(rp + 96  + half * 16);   // K 48-55 / 56-63

    const char*  cb_base = (const char*)cbbf + (size_t)q * KK * DD * 2;
    const float* c2q     = c2 + q * KK;

    float best[8];
    int   bidx[8];
    #pragma unroll
    for (int r = 0; r < 8; ++r) { best[r] = 3.4e38f; bidx[r] = 0; }

    // ---- sweep codebook in LDS-staged chunks of 128 codes ----
    for (int chunk = 0; chunk < KK / CHUNK_CODES; ++chunk) {
        // cooperative stage: 16 KB / 128 threads = 128 B (8 x b128) per thread
        {
            const u32x4* src = (const u32x4*)(cb_base +
                                (size_t)chunk * CHUNK_CODES * DD * 2);
            u32x4* dst = (u32x4*)scb;
            #pragma unroll
            for (int j = 0; j < 8; ++j)
                dst[threadIdx.x + 128 * j] = src[threadIdx.x + 128 * j];
        }
        __syncthreads();
        // overlap next chunk's global fetch with this chunk's WMMA sweep
        __builtin_prefetch(cb_base + (size_t)(chunk + 1) * CHUNK_CODES * DD * 2, 0, 0);

        #pragma unroll
        for (int tt = 0; tt < CHUNK_CODES / 16; ++tt) {   // 8 code tiles per chunk
            const int t = chunk * (CHUNK_CODES / 16) + tt;
            const char* cp = (const char*)scb +
                             ((size_t)(tt * 16 + lrow) * DD) * 2 + half * 16;
            Frag b0, b1;
            b0.u[0] = *(const u32x4*)(cp);                 // K 0-7   / 8-15
            b0.u[1] = *(const u32x4*)(cp + 32);            // K 16-23 / 24-31
            b1.u[0] = *(const u32x4*)(cp + 64);            // K 32-39 / 40-47
            b1.u[1] = *(const u32x4*)(cp + 96);            // K 48-55 / 56-63

            v8f c = {0.f, 0.f, 0.f, 0.f, 0.f, 0.f, 0.f, 0.f};
            c = __builtin_amdgcn_wmma_f32_16x16x32_bf16(false, a0.bf, false, b0.bf,
                                                        (short)0, c, false, false);
            c = __builtin_amdgcn_wmma_f32_16x16x32_bf16(false, a1.bf, false, b1.bf,
                                                        (short)0, c, false, false);

            // score = c^2 - 2*dot  (r^2 constant per row, irrelevant to argmin)
            float c2v = c2q[t * 16 + lrow];
            int   ci  = t * 16 + lrow;
            #pragma unroll
            for (int r = 0; r < 8; ++r) {
                float s = fmaf(-2.0f, c[r], c2v);          // row = rowbase + half*8 + r
                if (s < best[r]) { best[r] = s; bidx[r] = ci; }
            }
        }
        __syncthreads();
    }

    // ---- cross-lane argmin over the 16 codes striped in each half ----
    #pragma unroll
    for (int r = 0; r < 8; ++r) {
        float bv = best[r];
        int   bi = bidx[r];
        #pragma unroll
        for (int m = 1; m < 16; m <<= 1) {
            float ov = __shfl_xor(bv, m, 32);
            int   oi = __shfl_xor(bi, m, 32);
            if (ov < bv || (ov == bv && oi < bi)) { bv = ov; bi = oi; }
        }
        bidx[r] = bi;    // all lanes of a half now hold that half's row argmins
    }

    // gather winners: rows 0-7 from lane 0 (half0), rows 8-15 from lane 16 (half1)
    int rowidx[16];
    #pragma unroll
    for (int r = 0; r < 8; ++r) {
        rowidx[r]     = __shfl(bidx[r], 0, 32);
        rowidx[r + 8] = __shfl(bidx[r], 16, 32);
    }

    // ---- exact fp32 update: residual -= quant; outacc += quant; loss ----
    const float* cbq = cbf32 + (size_t)q * KK * DD;
    float lossacc = 0.0f;
    #pragma unroll
    for (int i = 0; i < 32; ++i) {                     // 16 rows x 64 dims / 32 lanes
        int rloc = i >> 1;                             // uniform per iteration
        int dim  = ((i & 1) << 5) + lane;
        int idx  = rowidx[rloc];
        float qv = cbq[(size_t)idx * DD + dim];
        size_t off = (size_t)(rowbase + rloc) * DD + dim;
        float rv = resid[off];
        float nr = rv - qv;
        resid[off] = nr;
        residbf_w[off] = __float2bfloat16(nr);         // next round's A operand
        outacc[off] += qv;
        float d = qv - rv;
        lossacc += d * d;
        if (lane == 0 && (i & 1) == 0)
            dout[IDX_AT + (size_t)(rowbase + rloc) * QQ + q] = (float)idx;
    }
    #pragma unroll
    for (int m = 1; m < 32; m <<= 1) lossacc += __shfl_xor(lossacc, m, 32);
    if (lane == 0) atomicAdd(&scal[1 + q], lossacc);
}

// ---------- finalize: transpose [b,n,d] -> [b,d,n]; emit scalars ----------
__global__ void k_fin(const float* __restrict__ outacc, const float* __restrict__ scal,
                      float* __restrict__ dout) {
    int gid = blockIdx.x * blockDim.x + threadIdx.x;   // over OUT_N
    if (gid < OUT_N) {
        int n = gid & (NN - 1);
        int d = (gid >> 12) & (DD - 1);
        int b = gid >> 18;
        dout[gid] = outacc[((size_t)b * NN + n) * DD + d];
    }
    if (gid == 0) {
        dout[KL_AT] = scal[0] / (float)(BB * NN);
        float ls = 0.0f;
        #pragma unroll
        for (int qq = 0; qq < QQ; ++qq) ls += scal[1 + qq];
        dout[LOSS_AT] = ls / ((float)(ROWS * DD) * (float)QQ);
    }
}

extern "C" void kernel_launch(void* const* d_in, const int* in_sizes, int n_in,
                              void* d_out, int out_size, void* d_ws, size_t ws_size,
                              hipStream_t stream) {
    (void)in_sizes; (void)n_in; (void)out_size; (void)ws_size;
    const float* x    = (const float*)d_in[0];
    const float* cb   = (const float*)d_in[1];
    const int*   seed = (const int*)d_in[2];
    float* dout = (float*)d_out;
    char*  ws   = (char*)d_ws;

    float*           resid   = (float*)(ws + WS_RESID);
    __hip_bfloat16*  residbf = (__hip_bfloat16*)(ws + WS_RESIDBF);
    float*           outacc  = (float*)(ws + WS_OUTACC);
    __hip_bfloat16*  cbbf    = (__hip_bfloat16*)(ws + WS_CBBF);
    float*           c2      = (float*)(ws + WS_C2);
    float*           scal    = (float*)(ws + WS_SCAL);

    k_zero<<<(ROWS * DD + 255) / 256, 256, 0, stream>>>(outacc, scal);
    k_cvt <<<(QQ * KK * DD + 255) / 256, 256, 0, stream>>>(cb, cbbf);
    k_c2  <<<(QQ * KK + 255) / 256, 256, 0, stream>>>(cb, c2);
    k_vae <<<(BB * DD * NN + 255) / 256, 256, 0, stream>>>(x, seed, resid, residbf, scal);

    for (int q = 0; q < QQ; ++q) {
        k_rvq<<<(ROWS / 16) / 4, 128, 0, stream>>>(residbf, resid, residbf, cbbf, cb,
                                                   c2, outacc, scal, dout, q);
    }
    k_fin<<<(OUT_N + 255) / 256, 256, 0, stream>>>(outacc, scal, dout);
}
